// VoxelResBackBone8x_77558519431891
// MI455X (gfx1250) — matrix operand order repaired and checked by reading, not compiled, not run on hardware
//
#include <hip/hip_runtime.h>
#include <cstdint>
#include <cstddef>

// ---------------------------------------------------------------------------
// MI455X (gfx1250) implementation of VoxelResBackBone8x.
// Dense implicit-GEMM 3D convs on bf16 NDHWC activations using
// V_WMMA_F32_16X16X32_BF16 (f32 accumulation). BN folded into weights,
// masked-ReLU / residual epilogues fused. Kernels templated on
// (Cin, kernel shape, N-tiles per wave); inner loop is software-pipelined
// (double-buffered A/B fragments) with branchless zero-page boundary handling.
// ---------------------------------------------------------------------------

#define WAVES_PER_BLOCK 8

typedef __attribute__((ext_vector_type(16))) __bf16 v16bf;
typedef __attribute__((ext_vector_type(8)))  float  v8f;
typedef __attribute__((ext_vector_type(4)))  unsigned int u32x4;

union Frag {
  v16bf v;
  unsigned short u[16];
  u32x4 q[2];
};

__device__ __forceinline__ float bf2f(unsigned short h) {
  union { unsigned int u; float f; } x; x.u = ((unsigned int)h) << 16; return x.f;
}
__device__ __forceinline__ unsigned short f2bf(float f) {
  union { float f; unsigned int u; } x; x.f = f;
  unsigned int r = x.u + 0x7FFFu + ((x.u >> 16) & 1u);   // round to nearest even
  return (unsigned short)(r >> 16);
}

struct ConvP {
  const unsigned short* act;    // input activations, bf16 NDHWC (Cin fastest)
  const unsigned short* wq;     // weights, bf16 [Cout][Kpad], K = tap*Cin+ci, BN-scale folded
  const float* bias;            // [Cout] BN shift
  const float* mask;            // [Dn*Hn*Wn] output-resolution mask
  const unsigned short* resid;  // residual (bf16 NDHWC at out res), mode 1 only
  const unsigned short* zpad;   // 64B zero page for OOB lanes
  unsigned short* out;          // bf16 NDHWC output
  float* outf;                  // f32 NCDHW output (mode 2)
  int Cout;
  int Di, Hi, Wi;               // input spatial dims
  int Dn, Hn, Wn;               // output spatial dims
  int sz, sy, sx, pz, py, px;   // stride / pad
  int mode;                     // 0 = mask*relu(v)          (subm & downsample convs)
                                // 1 = relu(v*mask + resid)  (residual block tail)
                                // 2 = mask*relu(v) -> f32 NCDHW (final conv_out)
  int mtiles, wtiles;
};

// One wave computes a 16-voxel x (NT*16)-output-channel tile via WMMA.
// A k-mapping (ISA 16-bit A 16x32): element i -> k = 16*(i>>3) + 8*half + (i&7)
// B k-mapping:                      element i -> k = 16*(i>>3) + 8*half + (i&7)
// D layout: VGPR r -> row M = r + 8*half, col N = lane & 15
template <int CIN, int KD, int KH, int KW, int NT>
__global__ __launch_bounds__(256) void conv3d_wmma(ConvP p) {
  constexpr int KHW  = KH * KW;
  constexpr int T    = KD * KH * KW;
  constexpr int KTOT = T * CIN;
  constexpr int KPAD = (KTOT + 31) & ~31;
  constexpr int CPT  = (CIN >= 32) ? (CIN / 32) : 1;      // K-chunks per tap
  constexpr int LCPT = (CPT == 4) ? 2 : (CPT == 2) ? 1 : 0;
  constexpr int NCH  = (CIN >= 32) ? (T * CPT)
                     : (CIN == 16) ? ((T + 1) / 2)
                                   : ((KTOT + 31) / 32);  // total K=32 chunks

  const int lane  = threadIdx.x & 31;
  const int wv    = threadIdx.x >> 5;
  const int mtile = blockIdx.x * WAVES_PER_BLOCK + wv;
  if (mtile >= p.mtiles) return;            // wave-uniform exit: EXEC all-1 for WMMA
  const int ntbase = blockIdx.y * NT;

  const int xt = mtile % p.wtiles;
  const int t2 = mtile / p.wtiles;
  const int yo = t2 % p.Hn;
  const int zo = t2 / p.Hn;

  const int half = lane >> 4;
  const int mrow = lane & 15;               // A-matrix row gathered by this lane
  const int ncol = lane & 15;               // B/D column owned by this lane
  const int xo_a = xt * 16 + mrow;

  const v8f z8 = {0.f, 0.f, 0.f, 0.f, 0.f, 0.f, 0.f, 0.f};
  v8f acc[NT];
  #pragma unroll
  for (int nt = 0; nt < NT; ++nt) acc[nt] = z8;

  const unsigned short* wrow[NT];
  #pragma unroll
  for (int nt = 0; nt < NT; ++nt)
    wrow[nt] = p.wq + (size_t)((ntbase + nt) * 16 + ncol) * KPAD + 8 * half;

  // ---- fragment loaders (branchless: OOB lanes read the zero page) ----
  auto loadA = [&](int cc, Frag& A) {
    if constexpr (CIN >= 32) {
      const int tap = cc >> LCPT;
      const int c0  = (cc & (CPT - 1)) * 32;
      const int tz = tap / KHW;
      const int tr = tap - tz * KHW;
      const int ty = tr / KW;
      const int tx = tr - ty * KW;
      const int zi = zo * p.sz + tz - p.pz;
      const int yi = yo * p.sy + ty - p.py;
      const int xi = xo_a * p.sx + tx - p.px;
      const bool ok = (unsigned)zi < (unsigned)p.Di &&
                      (unsigned)yi < (unsigned)p.Hi &&
                      (unsigned)xi < (unsigned)p.Wi;
      const unsigned short* ap = ok
          ? p.act + (size_t)((zi * p.Hi + yi) * p.Wi + xi) * CIN + c0 + 8 * half
          : p.zpad;
      A.q[0] = *(const u32x4*)(ap);          // k = c0 + 8*half .. +7   (16B aligned)
      A.q[1] = *(const u32x4*)(ap + 16);     // k = c0 + 16 + 8*half .. +7
    } else if constexpr (CIN == 16) {
      // Two taps per chunk: octet o covers tap = 2*cc + o, ci = 8*half..+7.
      #pragma unroll
      for (int o = 0; o < 2; ++o) {
        const int tap = 2 * cc + o;
        const int tz = tap / KHW;
        const int tr = tap - tz * KHW;
        const int ty = tr / KW;
        const int tx = tr - ty * KW;
        const int zi = zo * p.sz + tz - p.pz;
        const int yi = yo * p.sy + ty - p.py;
        const int xi = xo_a * p.sx + tx - p.px;
        const bool ok = (tap < T) &&
                        (unsigned)zi < (unsigned)p.Di &&
                        (unsigned)yi < (unsigned)p.Hi &&
                        (unsigned)xi < (unsigned)p.Wi;
        const unsigned short* ap = ok
            ? p.act + (size_t)((zi * p.Hi + yi) * p.Wi + xi) * CIN + 8 * half
            : p.zpad;
        A.q[o] = *(const u32x4*)(ap);
      }
    } else {
      // CIN == 4 (first conv only): per-element gather with zero K padding.
      const int kk = cc * 32;
      #pragma unroll
      for (int i = 0; i < 16; ++i) {
        const int k = 16 * (i >> 3) + 8 * half + (i & 7);
        const int g = kk + k;
        unsigned short e = 0;
        if (g < KTOT) {
          const int tap = g / CIN;
          const int ci  = g % CIN;
          const int tz = tap / KHW;
          const int tr = tap - tz * KHW;
          const int ty = tr / KW;
          const int tx = tr - ty * KW;
          const int zi = zo * p.sz + tz - p.pz;
          const int yi = yo * p.sy + ty - p.py;
          const int xi = xo_a * p.sx + tx - p.px;
          if ((unsigned)zi < (unsigned)p.Di && (unsigned)yi < (unsigned)p.Hi &&
              (unsigned)xi < (unsigned)p.Wi) {
            e = p.act[(size_t)((zi * p.Hi + yi) * p.Wi + xi) * CIN + ci];
          }
        }
        A.u[i] = e;
      }
    }
  };

  auto loadB = [&](int cc, Frag* B) {
    #pragma unroll
    for (int nt = 0; nt < NT; ++nt) {
      const unsigned short* wp = wrow[nt] + cc * 32;
      B[nt].q[0] = *(const u32x4*)(wp);
      B[nt].q[1] = *(const u32x4*)(wp + 16);
    }
    __builtin_prefetch(wrow[0] + cc * 32 + 64, 0, 3);   // global_prefetch_b8, no branch
  };

  auto domma = [&](const Frag& A, const Frag* B) {
    #pragma unroll
    for (int nt = 0; nt < NT; ++nt)
      acc[nt] = __builtin_amdgcn_wmma_f32_16x16x32_bf16(
          false, A.v, false, B[nt].v, (short)0, acc[nt], false, false);
  };

  // ---- software-pipelined K loop (double-buffered fragments) ----
  Frag A0, A1, B0[NT], B1[NT];
  loadA(0, A0); loadB(0, B0);
  int cc = 0;
  #pragma unroll 1
  for (; cc + 2 < NCH; cc += 2) {
    loadA(cc + 1, A1); loadB(cc + 1, B1);
    domma(A0, B0);
    loadA(cc + 2, A0); loadB(cc + 2, B0);
    domma(A1, B1);
  }
  if (cc + 1 < NCH) {            // two chunks remain
    loadA(cc + 1, A1); loadB(cc + 1, B1);
    domma(A0, B0);
    domma(A1, B1);
  } else {                       // one chunk remains
    domma(A0, B0);
  }

  // ---- Fused epilogue ----
  #pragma unroll
  for (int nt = 0; nt < NT; ++nt) {
    const int   cout = (ntbase + nt) * 16 + ncol;
    const float bb   = p.bias[cout];
    #pragma unroll
    for (int r = 0; r < 8; ++r) {
      const int md = r + 8 * half;
      const int xo = xt * 16 + md;
      if (xo >= p.Wn) continue;
      const size_t vox = ((size_t)zo * p.Hn + yo) * p.Wn + xo;
      float v = acc[nt][r] + bb;
      if (p.mode == 1) {
        v = v * p.mask[vox] + bf2f(p.resid[vox * p.Cout + cout]);
        v = fmaxf(v, 0.f);
        p.out[vox * p.Cout + cout] = f2bf(v);
      } else if (p.mode == 2) {
        v = fmaxf(v, 0.f) * p.mask[vox];
        p.outf[(((size_t)cout * p.Dn + zo) * p.Hn + yo) * p.Wn + xo] = v;
      } else {
        v = fmaxf(v, 0.f) * p.mask[vox];
        p.out[vox * p.Cout + cout] = f2bf(v);
      }
    }
  }
}

// ---------------------------------------------------------------------------
__global__ void fill_zero_u32(unsigned int* p, long long n) {
  long long i = (long long)blockIdx.x * blockDim.x + threadIdx.x;
  long long stride = (long long)gridDim.x * blockDim.x;
  for (; i < n; i += stride) p[i] = 0u;
}

__global__ void scatter_voxels(const float* feat, const int* coords,
                               unsigned short* act4, float* mask, int n,
                               int Dd, int Hh, int Ww) {
  int i = blockIdx.x * blockDim.x + threadIdx.x;
  if (i >= n) return;
  int z = coords[i * 4 + 1], y = coords[i * 4 + 2], x = coords[i * 4 + 3];
  if ((unsigned)z >= (unsigned)Dd || (unsigned)y >= (unsigned)Hh ||
      (unsigned)x >= (unsigned)Ww) return;
  size_t vox = ((size_t)z * Hh + y) * Ww + x;
  #pragma unroll
  for (int c = 0; c < 4; ++c) act4[vox * 4 + c] = f2bf(feat[i * 4 + c]);
  mask[vox] = 1.f;
}

// Repack OIDHW f32 weights -> [Cout][Kpad] bf16 with BN scale folded; bias = BN shift.
__global__ void repack_w(const float* w, const float* gamma, const float* beta,
                         const float* mean, const float* var,
                         unsigned short* wq, float* bias,
                         int Cout, int Cin, int T, int Ktot, int Kpad) {
  int idx = blockIdx.x * blockDim.x + threadIdx.x;
  int total = Cout * Kpad;
  if (idx >= total) return;
  int co = idx / Kpad, gp = idx - co * Kpad;
  float sc = gamma[co] * rsqrtf(var[co] + 1e-3f);
  unsigned short o = 0;
  if (gp < Ktot) {
    int tap = gp / Cin, ci = gp - tap * Cin;
    o = f2bf(w[((size_t)co * Cin + ci) * T + tap] * sc);
  }
  wq[idx] = o;
  if (gp == 0) bias[co] = beta[co] - mean[co] * sc;
}

// mask_out = (conv(mask_in, ones, stride, pad) > 0)
__global__ void mask_down(const float* mi, float* mo,
                          int Di, int Hi, int Wi, int Dn, int Hn, int Wn,
                          int kd, int kh, int kw, int sz, int sy, int sx,
                          int pz, int py, int px) {
  int idx = blockIdx.x * blockDim.x + threadIdx.x;
  int tot = Dn * Hn * Wn;
  if (idx >= tot) return;
  int x = idx % Wn; int t = idx / Wn; int y = t % Hn; int z = t / Hn;
  float any = 0.f;
  for (int tz = 0; tz < kd; ++tz) {
    int zi = z * sz + tz - pz; if ((unsigned)zi >= (unsigned)Di) continue;
    for (int ty = 0; ty < kh; ++ty) {
      int yi = y * sy + ty - py; if ((unsigned)yi >= (unsigned)Hi) continue;
      for (int tx = 0; tx < kw; ++tx) {
        int xi = x * sx + tx - px; if ((unsigned)xi >= (unsigned)Wi) continue;
        if (mi[((size_t)zi * Hi + yi) * Wi + xi] > 0.f) any = 1.f;
      }
    }
  }
  mo[idx] = any;
}

// ---------------------------------------------------------------------------
extern "C" void kernel_launch(void* const* d_in, const int* in_sizes, int n_in,
                              void* d_out, int out_size, void* d_ws, size_t ws_size,
                              hipStream_t stream) {
  (void)n_in; (void)out_size; (void)ws_size;
  char* ws = (char*)d_ws;
  size_t off = 0;
  auto alloc = [&](size_t bytes) -> char* {
    char* p = ws + off;
    off = (off + bytes + 255) & ~(size_t)255;
    return p;
  };

  const int D0 = 41, H0 = 160, W0 = 160;
  const int D1 = 21, H1 = 80,  W1 = 80;
  const int D2 = 11, H2 = 40,  W2 = 40;
  const int D3 = 5,  H3 = 20,  W3 = 20;
  const int D4 = 2,  H4 = 20,  W4 = 20;

  unsigned short* in4  = (unsigned short*)alloc((size_t)D0 * H0 * W0 * 4 * 2);
  float* mask0 = (float*)alloc((size_t)D0 * H0 * W0 * 4);
  float* mask1 = (float*)alloc((size_t)D1 * H1 * W1 * 4);
  float* mask2 = (float*)alloc((size_t)D2 * H2 * W2 * 4);
  float* mask3 = (float*)alloc((size_t)D3 * H3 * W3 * 4);
  float* mask4 = (float*)alloc((size_t)D4 * H4 * W4 * 4);
  unsigned short* zpad = (unsigned short*)alloc(256);   // zero page for OOB lanes
  size_t bufBytes = (size_t)D0 * H0 * W0 * 16 * 2;      // largest activation (bf16)
  unsigned short* bufA = (unsigned short*)alloc(bufBytes);
  unsigned short* bufB = (unsigned short*)alloc(bufBytes);
  unsigned short* bufC = (unsigned short*)alloc(bufBytes);

  struct WQ { const unsigned short* w; const float* b; };
  // params flattened depth-first in dict/list insertion order; each conv layer is
  // w at d_in[i], then bn gamma/beta/mean/var at d_in[i+1..i+4].
  auto repack = [&](int wIdx, int Cin, int Cout, int T) -> WQ {
    int Ktot = T * Cin;
    int Kpad = (Ktot + 31) & ~31;
    unsigned short* wq = (unsigned short*)alloc((size_t)Cout * Kpad * 2);
    float* bias = (float*)alloc((size_t)Cout * 4);
    int tot = Cout * Kpad;
    repack_w<<<dim3((tot + 255) / 256), dim3(256), 0, stream>>>(
        (const float*)d_in[wIdx], (const float*)d_in[wIdx + 1],
        (const float*)d_in[wIdx + 2], (const float*)d_in[wIdx + 3],
        (const float*)d_in[wIdx + 4], wq, bias, Cout, Cin, T, Ktot, Kpad);
    return WQ{wq, bias};
  };

  auto conv = [&](const unsigned short* act, WQ w, const float* mask,
                  const unsigned short* resid, unsigned short* outp, float* outf,
                  int Cin, int Cout, int Di, int Hi, int Wi,
                  int Dn, int Hn, int Wn,
                  int kd, int sz, int sy, int sx,
                  int pz, int py, int px, int mode) {
    ConvP p;
    p.act = act; p.wq = w.w; p.bias = w.b; p.mask = mask; p.resid = resid;
    p.zpad = zpad;
    p.out = outp; p.outf = outf;
    p.Cout = Cout;
    p.Di = Di; p.Hi = Hi; p.Wi = Wi; p.Dn = Dn; p.Hn = Hn; p.Wn = Wn;
    p.sz = sz; p.sy = sy; p.sx = sx; p.pz = pz; p.py = py; p.px = px;
    p.mode = mode;
    p.wtiles = (Wn + 15) / 16;
    p.mtiles = Dn * Hn * p.wtiles;
    const int NT = (Cout >= 32) ? 2 : 1;
    dim3 grid((p.mtiles + WAVES_PER_BLOCK - 1) / WAVES_PER_BLOCK,
              Cout / (16 * NT));
    dim3 blk(WAVES_PER_BLOCK * 32);
    if (kd == 1) {  // marker for the (3,1,1) conv_out layer
      conv3d_wmma<128, 3, 1, 1, 2><<<grid, blk, 0, stream>>>(p);
    } else if (Cin == 4) {
      conv3d_wmma<4, 3, 3, 3, 1><<<grid, blk, 0, stream>>>(p);
    } else if (Cin == 16 && NT == 1) {
      conv3d_wmma<16, 3, 3, 3, 1><<<grid, blk, 0, stream>>>(p);
    } else if (Cin == 16) {
      conv3d_wmma<16, 3, 3, 3, 2><<<grid, blk, 0, stream>>>(p);
    } else if (Cin == 32) {
      conv3d_wmma<32, 3, 3, 3, 2><<<grid, blk, 0, stream>>>(p);
    } else if (Cin == 64) {
      conv3d_wmma<64, 3, 3, 3, 2><<<grid, blk, 0, stream>>>(p);
    } else {
      conv3d_wmma<128, 3, 3, 3, 2><<<grid, blk, 0, stream>>>(p);
    }
  };

  // ---- weight preprocessing (independent of activations) ----
  WQ w_in = repack(3,   4,  16, 27);
  WQ w10a = repack(8,  16,  16, 27), w10b = repack(13, 16, 16, 27);
  WQ w11a = repack(18, 16,  16, 27), w11b = repack(23, 16, 16, 27);
  WQ w2d  = repack(28, 16,  32, 27);
  WQ w20a = repack(33, 32,  32, 27), w20b = repack(38, 32, 32, 27);
  WQ w21a = repack(43, 32,  32, 27), w21b = repack(48, 32, 32, 27);
  WQ w3d  = repack(53, 32,  64, 27);
  WQ w30a = repack(58, 64,  64, 27), w30b = repack(63, 64, 64, 27);
  WQ w31a = repack(68, 64,  64, 27), w31b = repack(73, 64, 64, 27);
  WQ w4d  = repack(78, 64, 128, 27);
  WQ w40a = repack(83, 128, 128, 27), w40b = repack(88, 128, 128, 27);
  WQ w41a = repack(93, 128, 128, 27), w41b = repack(98, 128, 128, 27);
  WQ wout = repack(103, 128, 128, 3);

  // ---- zero-init + scatter voxels into dense bf16 NDHWC grid + mask0 ----
  long long in4u32 = ((long long)D0 * H0 * W0 * 4 * 2 + 3) / 4;
  fill_zero_u32<<<dim3(2048), dim3(256), 0, stream>>>((unsigned int*)in4, in4u32);
  fill_zero_u32<<<dim3(2048), dim3(256), 0, stream>>>((unsigned int*)mask0,
                                                      (long long)D0 * H0 * W0);
  fill_zero_u32<<<dim3(1), dim3(64), 0, stream>>>((unsigned int*)zpad, 64);
  int nvox = in_sizes[0] / 4;
  scatter_voxels<<<dim3((nvox + 255) / 256), dim3(256), 0, stream>>>(
      (const float*)d_in[0], (const int*)d_in[1], in4, mask0, nvox, D0, H0, W0);

  // ---- conv_input + conv1 (full res, 16 ch) ----
  conv(in4,  w_in,  mask0, nullptr, bufA, nullptr, 4, 16, D0,H0,W0, D0,H0,W0, 3, 1,1,1, 1,1,1, 0);
  conv(bufA, w10a, mask0, nullptr, bufB, nullptr, 16,16, D0,H0,W0, D0,H0,W0, 3, 1,1,1, 1,1,1, 0);
  conv(bufB, w10b, mask0, bufA,    bufC, nullptr, 16,16, D0,H0,W0, D0,H0,W0, 3, 1,1,1, 1,1,1, 1);
  conv(bufC, w11a, mask0, nullptr, bufA, nullptr, 16,16, D0,H0,W0, D0,H0,W0, 3, 1,1,1, 1,1,1, 0);
  conv(bufA, w11b, mask0, bufC,    bufB, nullptr, 16,16, D0,H0,W0, D0,H0,W0, 3, 1,1,1, 1,1,1, 1);

  // ---- stage 2 (21,80,80, 32 ch) ----
  mask_down<<<dim3((D1*H1*W1 + 255)/256), dim3(256), 0, stream>>>(
      mask0, mask1, D0,H0,W0, D1,H1,W1, 3,3,3, 2,2,2, 1,1,1);
  conv(bufB, w2d,  mask1, nullptr, bufA, nullptr, 16,32, D0,H0,W0, D1,H1,W1, 3, 2,2,2, 1,1,1, 0);
  conv(bufA, w20a, mask1, nullptr, bufB, nullptr, 32,32, D1,H1,W1, D1,H1,W1, 3, 1,1,1, 1,1,1, 0);
  conv(bufB, w20b, mask1, bufA,    bufC, nullptr, 32,32, D1,H1,W1, D1,H1,W1, 3, 1,1,1, 1,1,1, 1);
  conv(bufC, w21a, mask1, nullptr, bufA, nullptr, 32,32, D1,H1,W1, D1,H1,W1, 3, 1,1,1, 1,1,1, 0);
  conv(bufA, w21b, mask1, bufC,    bufB, nullptr, 32,32, D1,H1,W1, D1,H1,W1, 3, 1,1,1, 1,1,1, 1);

  // ---- stage 3 (11,40,40, 64 ch) ----
  mask_down<<<dim3((D2*H2*W2 + 255)/256), dim3(256), 0, stream>>>(
      mask1, mask2, D1,H1,W1, D2,H2,W2, 3,3,3, 2,2,2, 1,1,1);
  conv(bufB, w3d,  mask2, nullptr, bufA, nullptr, 32,64, D1,H1,W1, D2,H2,W2, 3, 2,2,2, 1,1,1, 0);
  conv(bufA, w30a, mask2, nullptr, bufB, nullptr, 64,64, D2,H2,W2, D2,H2,W2, 3, 1,1,1, 1,1,1, 0);
  conv(bufB, w30b, mask2, bufA,    bufC, nullptr, 64,64, D2,H2,W2, D2,H2,W2, 3, 1,1,1, 1,1,1, 1);
  conv(bufC, w31a, mask2, nullptr, bufA, nullptr, 64,64, D2,H2,W2, D2,H2,W2, 3, 1,1,1, 1,1,1, 0);
  conv(bufA, w31b, mask2, bufC,    bufB, nullptr, 64,64, D2,H2,W2, D2,H2,W2, 3, 1,1,1, 1,1,1, 1);

  // ---- stage 4 (5,20,20, 128 ch), pad (0,1,1) on the downsample ----
  mask_down<<<dim3((D3*H3*W3 + 255)/256), dim3(256), 0, stream>>>(
      mask2, mask3, D2,H2,W2, D3,H3,W3, 3,3,3, 2,2,2, 0,1,1);
  conv(bufB, w4d,  mask3, nullptr, bufA, nullptr, 64,128, D2,H2,W2, D3,H3,W3, 3, 2,2,2, 0,1,1, 0);
  conv(bufA, w40a, mask3, nullptr, bufB, nullptr, 128,128, D3,H3,W3, D3,H3,W3, 3, 1,1,1, 1,1,1, 0);
  conv(bufB, w40b, mask3, bufA,    bufC, nullptr, 128,128, D3,H3,W3, D3,H3,W3, 3, 1,1,1, 1,1,1, 1);
  conv(bufC, w41a, mask3, nullptr, bufA, nullptr, 128,128, D3,H3,W3, D3,H3,W3, 3, 1,1,1, 1,1,1, 0);
  conv(bufA, w41b, mask3, bufC,    bufB, nullptr, 128,128, D3,H3,W3, D3,H3,W3, 3, 1,1,1, 1,1,1, 1);

  // ---- conv_out: (3,1,1) stride (2,1,1) pad 0 -> f32 NCDHW into d_out ----
  mask_down<<<dim3((D4*H4*W4 + 255)/256), dim3(256), 0, stream>>>(
      mask3, mask4, D3,H3,W3, D4,H4,W4, 3,1,1, 2,1,1, 0,0,0);
  // kd==1 marker selects the <128,3,1,1,2> instantiation
  conv(bufB, wout, mask4, nullptr, nullptr, (float*)d_out, 128,128,
       D3,H3,W3, D4,H4,W4, 1, 2,1,1, 0,0,0, 2);
}
